// WindowAttention_83674552861036
// MI455X (gfx1250) — matrix operand order repaired and checked by reading, not compiled, not run on hardware
//
#include <hip/hip_runtime.h>
#include <hip/hip_bf16.h>
#include <hip/hip_fp16.h>

// ---------------------------------------------------------------------------
// Window attention (Swin) for MI455X / gfx1250, wave32 + WMMA f16->f32.
//   B=16, L=4096 (H=W=64), C=768, heads=12, head_dim=64, window 8x8 (64 tok).
// Pipeline: partition+f16 -> QKV GEMMs (WMMA, double-buffered async LDS
// staging) -> per-(window,head) attention (WMMA, S^T trick for in-lane
// softmax) -> output GEMM with fused window reverse.  Workspace ~408 MB.
// ---------------------------------------------------------------------------

typedef __attribute__((ext_vector_type(16))) _Float16 v16h;
typedef __attribute__((ext_vector_type(8)))  float    v8f;
typedef __attribute__((ext_vector_type(4)))  int      v4i;

#define C_DIM 768
#define NHEAD 12
#define HDIM  64
#define NTOK  64      // tokens per window
#define NWIN  1024    // 16 * 8 * 8 windows
#define TTOT  65536   // NWIN * NTOK window-ordered tokens

union Frag16 { v16h h; unsigned int u[8]; };
union FragF  { v8f  f; float s[8]; };

// ---- CDNA5 async global->LDS copy (ASYNCcnt) with safe fallbacks ----------
#if defined(__HIP_DEVICE_COMPILE__) && __has_builtin(__builtin_amdgcn_global_load_async_to_lds_b128)
#define HAVE_ASYNC_LDS 1
#else
#define HAVE_ASYNC_LDS 0
#endif

__device__ __forceinline__ void cp16(_Float16* l, const _Float16* g) {
#if HAVE_ASYNC_LDS
  // 16B per lane, memory -> LDS directly, no VGPR staging. Tracked by ASYNCcnt.
  // Prototype (from clang): (v4i AS1* gsrc, v4i AS3* ldst, imm offset, imm cpol)
  __builtin_amdgcn_global_load_async_to_lds_b128(
      (__attribute__((address_space(1))) v4i*)g,
      (__attribute__((address_space(3))) v4i*)l, 0, 0);
#else
  *reinterpret_cast<uint4*>(l) = *reinterpret_cast<const uint4*>(g);
#endif
}

template <int N>
__device__ __forceinline__ void wait_cp() {
#if HAVE_ASYNC_LDS
#if __has_builtin(__builtin_amdgcn_s_wait_asynccnt)
  __builtin_amdgcn_s_wait_asynccnt(N);
#else
  asm volatile("s_wait_asynccnt %0" ::"n"(N) : "memory");
#endif
#endif
}

__device__ __forceinline__ unsigned int pack2(float a, float b) {
  union { _Float16 h[2]; unsigned int u; } x;
  x.h[0] = (_Float16)a; x.h[1] = (_Float16)b;
  return x.u;
}

// ---- window-ordered token t -> flat (B,L) token ---------------------------
__device__ __forceinline__ int win_to_flat(int t) {
  int w = t >> 6, n = t & 63;
  int b = w >> 6, wl = w & 63;
  int r = ((wl >> 3) << 3) + (n >> 3);
  int c = ((wl & 7) << 3) + (n & 7);
  return b * 4096 + r * 64 + c;
}

// ---------------------------------------------------------------------------
// Kernel 1: window partition + f32 -> f16 convert.  grid = TTOT, block = 256
// ---------------------------------------------------------------------------
__global__ void prep_xw(const float* __restrict__ hs, _Float16* __restrict__ xw) {
  int t = blockIdx.x;
  const float*  src = hs + (size_t)win_to_flat(t) * C_DIM;
  _Float16*     dst = xw + (size_t)t * C_DIM;
  for (int i = threadIdx.x; i < C_DIM; i += blockDim.x)
    dst[i] = (_Float16)src[i];
}

// ---------------------------------------------------------------------------
// Kernel 2: elementwise f32 -> f16 (weights, layout preserved: [co][ci])
// ---------------------------------------------------------------------------
__global__ void cvt_f32_f16(const float* __restrict__ s, _Float16* __restrict__ d, int n) {
  for (int i = blockIdx.x * blockDim.x + threadIdx.x; i < n; i += gridDim.x * blockDim.x)
    d[i] = (_Float16)s[i];
}

// ---------------------------------------------------------------------------
// Kernels 3/5: tiled WMMA GEMM  Out = A(f16, MxK row-major) * W^T + bias
//   A: [TTOT][768] f16, W16: [768][768] f16 ([c_out][c_in], i.e. B = K x N
//   with contiguous K pairs per output column -> natural B-fragment loads).
//   Block tile 128x64, BK=32, 8 waves (4x2), each wave 32x32 (2x2 wmma tiles).
//   Double-buffered async global->LDS staging pipelined against WMMA.
//   MODE 0: row-major f16 store (Q,K).  MODE 1: per-window transposed V store.
// ---------------------------------------------------------------------------
#define LDP 40   // padded LDS row (f16 units): 40*2B = 20 banks -> conflict-free

// issue the 3 x 16B async copies this thread owns for one (buffer, k0) panel
__device__ __forceinline__ void gemm_stage(
    _Float16* As, _Float16* Bs, const _Float16* A, const _Float16* Bw,
    int bm, int bn, int k0, int tid) {
  for (int it = 0; it < 2; ++it) {
    int u = tid + it * 256;
    int row = u >> 2, seg = u & 3;
    cp16(&As[row * LDP + seg * 8], A + (size_t)(bm + row) * C_DIM + k0 + seg * 8);
  }
  {
    int row = tid >> 2, seg = tid & 3;
    cp16(&Bs[row * LDP + seg * 8], Bw + (size_t)(bn + row) * C_DIM + k0 + seg * 8);
  }
}

template <int MODE>
__global__ __launch_bounds__(256) void gemm_proj(
    const _Float16* __restrict__ A, const _Float16* __restrict__ Bw,
    const float* __restrict__ bias, _Float16* __restrict__ Out) {
  __shared__ _Float16 As[2][128 * LDP];
  __shared__ _Float16 Bs[2][64 * LDP];

  const int tid  = threadIdx.x;
  const int lane = tid & 31;
  const int sub  = lane >> 4;     // half-wave
  const int l15  = lane & 15;
  const int wave = tid >> 5;
  const int wm   = wave & 3;      // 0..3 -> 32-row strip
  const int wn   = wave >> 2;     // 0..1 -> 32-col strip
  const int bm   = blockIdx.x * 128;
  const int bn   = blockIdx.y * 64;

  FragF acc[2][2];
  for (int i = 0; i < 2; ++i)
    for (int j = 0; j < 2; ++j)
      for (int r = 0; r < 8; ++r) acc[i][j].s[r] = 0.0f;

  gemm_stage(As[0], Bs[0], A, Bw, bm, bn, 0, tid);   // prologue: buffer 0

  int cur = 0;
  for (int k0 = 0; k0 < C_DIM; k0 += 32) {
    if (k0 + 32 < C_DIM) {                           // prefetch next panel
      gemm_stage(As[cur ^ 1], Bs[cur ^ 1], A, Bw, bm, bn, k0 + 32, tid);
      wait_cp<3>();                                  // current panel landed
    } else {
      wait_cp<0>();
    }
    __syncthreads();

    Frag16 af[2], bf[2];
    for (int i = 0; i < 2; ++i) {                    // A frag: lane=M, vgprs=K
      const _Float16* ap = &As[cur][(wm * 32 + i * 16 + l15) * LDP + sub * 8];
      for (int v = 0; v < 8; ++v) {
        int kk = (v < 4) ? (2 * v) : (16 + 2 * (v - 4));
        af[i].u[v] = *reinterpret_cast<const unsigned int*>(ap + kk);
      }
    }
    for (int j = 0; j < 2; ++j) {                    // B frag: lane=N, vgprs=K
      const _Float16* bp = &Bs[cur][(wn * 32 + j * 16 + l15) * LDP + sub * 8];
      for (int v = 0; v < 8; ++v)
        bf[j].u[v] = *reinterpret_cast<const unsigned int*>(bp + ((v < 4) ? (2 * v) : (16 + 2 * (v - 4))));
    }
    for (int i = 0; i < 2; ++i)
      for (int j = 0; j < 2; ++j)
        acc[i][j].f = __builtin_amdgcn_wmma_f32_16x16x32_f16(
            false, af[i].h, false, bf[j].h, (short)0, acc[i][j].f, false, false);
    __syncthreads();                                 // done reading `cur`
    cur ^= 1;
  }

  // epilogue: D layout -> lane = N(col), vgpr r = M row (+8 for upper half)
  for (int i = 0; i < 2; ++i) {
    int rowbase = bm + wm * 32 + i * 16 + sub * 8;
    for (int j = 0; j < 2; ++j) {
      int col = bn + wn * 32 + j * 16 + l15;
      float bb = bias[col];
      if (MODE == 0) {
        for (int r = 0; r < 8; ++r)
          Out[(size_t)(rowbase + r) * C_DIM + col] = (_Float16)(acc[i][j].s[r] + bb);
      } else {
        // V transposed per window: Vt[w][col][token]  (rows stay in one window)
        int w = rowbase >> 6, n0 = rowbase & 63;
        _Float16* dst = Out + (size_t)w * (C_DIM * NTOK) + (size_t)col * NTOK + n0;
        for (int r = 0; r < 8; r += 2)
          *reinterpret_cast<unsigned int*>(dst + r) =
              pack2(acc[i][j].s[r] + bb, acc[i][j].s[r + 1] + bb);
      }
    }
  }
}

// ---------------------------------------------------------------------------
// Kernel 4: attention, one wave per (window, head).
//   S^T = K(64xd) x Q^T(dx64): softmax over keys m is 32 in-lane values plus
//   one shfl_xor(16); P^T accumulators repack *in-lane* into A-fragments of P
//   for ctx = P x V (V pre-transposed).  block = 256 (8 waves).
// ---------------------------------------------------------------------------
__global__ __launch_bounds__(256) void attn(
    const _Float16* __restrict__ Q, const _Float16* __restrict__ K,
    const _Float16* __restrict__ Vt, _Float16* __restrict__ Ctx) {
  const int lane = threadIdx.x & 31;
  const int sub  = lane >> 4;
  const int l15  = lane & 15;
  const int task = blockIdx.x * 8 + (threadIdx.x >> 5);   // 0..12287
  const int w = task / NHEAD, h = task % NHEAD;

  const _Float16* Qw = Q  + (size_t)w * NTOK * C_DIM + h * HDIM;
  const _Float16* Kw = K  + (size_t)w * NTOK * C_DIM + h * HDIM;
  const _Float16* Vw = Vt + (size_t)w * (C_DIM * NTOK) + (size_t)(h * HDIM) * NTOK;
  _Float16*       Cw = Ctx + (size_t)w * NTOK * C_DIM + h * HDIM;

  // Preload K as A-fragments of S^T: 4 m-tiles x 2 d-chunks (K-dim = 64)
  Frag16 ak[4][2];
  for (int mt = 0; mt < 4; ++mt) {
    const _Float16* kp = Kw + (size_t)(mt * 16 + l15) * C_DIM;
    for (int dc = 0; dc < 2; ++dc)
      for (int v = 0; v < 8; ++v) {
        int kk = dc * 32 + ((v < 4) ? 2 * v : 16 + 2 * (v - 4)) + sub * 8;
        ak[mt][dc].u[v] = *reinterpret_cast<const unsigned int*>(kp + kk);
      }
  }

  for (int nt = 0; nt < 4; ++nt) {                 // 16 queries per pass
    Frag16 bq[2];                                  // Q^T B-fragments
    {
      const _Float16* qp = Qw + (size_t)(nt * 16 + l15) * C_DIM;
      for (int dc = 0; dc < 2; ++dc)
        for (int v = 0; v < 8; ++v)
          bq[dc].u[v] = *reinterpret_cast<const unsigned int*>(qp + dc * 32 + sub * 8 + ((v < 4) ? (2 * v) : (16 + 2 * (v - 4))));
    }

    FragF st[4];
    for (int mt = 0; mt < 4; ++mt)
      for (int r = 0; r < 8; ++r) st[mt].s[r] = 0.0f;
    for (int mt = 0; mt < 4; ++mt)
      for (int dc = 0; dc < 2; ++dc)
        st[mt].f = __builtin_amdgcn_wmma_f32_16x16x32_f16(
            false, ak[mt][dc].h, false, bq[dc].h, (short)0, st[mt].f, false, false);

    // softmax over m (rows of S^T) for each query column n = lane&15
    const float scale = 0.125f;                    // 1/sqrt(64)
    float mx = -3.0e38f;
    for (int mt = 0; mt < 4; ++mt)
      for (int r = 0; r < 8; ++r) {
        st[mt].s[r] *= scale;
        mx = fmaxf(mx, st[mt].s[r]);
      }
    mx = fmaxf(mx, __shfl_xor(mx, 16, 32));
    float sum = 0.0f;
    for (int mt = 0; mt < 4; ++mt)
      for (int r = 0; r < 8; ++r) {
        float e = __expf(st[mt].s[r] - mx);
        st[mt].s[r] = e;
        sum += e;
      }
    sum += __shfl_xor(sum, 16, 32);
    float inv = 1.0f / sum;

    // P^T (D layout) -> A-fragments of P, purely in-lane
    Frag16 pa[2];
    for (int c2 = 0; c2 < 2; ++c2)
      for (int v = 0; v < 8; ++v) {
        int mt = c2 * 2 + ((v < 4) ? 0 : 1);
        int r  = (v < 4) ? 2 * v : 2 * (v - 4);
        pa[c2].u[v] = pack2(st[mt].s[r] * inv, st[mt].s[r + 1] * inv);
      }

    // ctx[nt-rows][d] = P x V   (V transposed -> contiguous B-frag loads)
    for (int dt = 0; dt < 4; ++dt) {
      FragF ca;
      for (int r = 0; r < 8; ++r) ca.s[r] = 0.0f;
      for (int c2 = 0; c2 < 2; ++c2) {
        Frag16 vb;
        const _Float16* vp = Vw + (size_t)(dt * 16 + l15) * NTOK + c2 * 32 + sub * 8;
        for (int v = 0; v < 8; ++v)
          vb.u[v] = *reinterpret_cast<const unsigned int*>(vp + ((v < 4) ? (2 * v) : (16 + 2 * (v - 4))));
        ca.f = __builtin_amdgcn_wmma_f32_16x16x32_f16(
            false, pa[c2].h, false, vb.h, (short)0, ca.f, false, false);
      }
      int nb = nt * 16 + sub * 8;
      for (int r = 0; r < 8; ++r)
        Cw[(size_t)(nb + r) * C_DIM + dt * 16 + l15] = (_Float16)ca.s[r];
    }
  }
}

// ---------------------------------------------------------------------------
// Kernel 5: output projection, f32 epilogue with fused window reverse.
// ---------------------------------------------------------------------------
__global__ __launch_bounds__(256) void gemm_out(
    const _Float16* __restrict__ A, const _Float16* __restrict__ Bw,
    const float* __restrict__ bias, float* __restrict__ Out) {
  __shared__ _Float16 As[2][128 * LDP];
  __shared__ _Float16 Bs[2][64 * LDP];

  const int tid  = threadIdx.x;
  const int lane = tid & 31;
  const int sub  = lane >> 4;
  const int l15  = lane & 15;
  const int wave = tid >> 5;
  const int wm   = wave & 3;
  const int wn   = wave >> 2;
  const int bm   = blockIdx.x * 128;
  const int bn   = blockIdx.y * 64;

  FragF acc[2][2];
  for (int i = 0; i < 2; ++i)
    for (int j = 0; j < 2; ++j)
      for (int r = 0; r < 8; ++r) acc[i][j].s[r] = 0.0f;

  gemm_stage(As[0], Bs[0], A, Bw, bm, bn, 0, tid);

  int cur = 0;
  for (int k0 = 0; k0 < C_DIM; k0 += 32) {
    if (k0 + 32 < C_DIM) {
      gemm_stage(As[cur ^ 1], Bs[cur ^ 1], A, Bw, bm, bn, k0 + 32, tid);
      wait_cp<3>();
    } else {
      wait_cp<0>();
    }
    __syncthreads();

    Frag16 af[2], bf[2];
    for (int i = 0; i < 2; ++i) {
      const _Float16* ap = &As[cur][(wm * 32 + i * 16 + l15) * LDP + sub * 8];
      for (int v = 0; v < 8; ++v) {
        int kk = (v < 4) ? (2 * v) : (16 + 2 * (v - 4));
        af[i].u[v] = *reinterpret_cast<const unsigned int*>(ap + kk);
      }
    }
    for (int j = 0; j < 2; ++j) {
      const _Float16* bp = &Bs[cur][(wn * 32 + j * 16 + l15) * LDP + sub * 8];
      for (int v = 0; v < 8; ++v)
        bf[j].u[v] = *reinterpret_cast<const unsigned int*>(bp + ((v < 4) ? (2 * v) : (16 + 2 * (v - 4))));
    }
    for (int i = 0; i < 2; ++i)
      for (int j = 0; j < 2; ++j)
        acc[i][j].f = __builtin_amdgcn_wmma_f32_16x16x32_f16(
            false, af[i].h, false, bf[j].h, (short)0, acc[i][j].f, false, false);
    __syncthreads();
    cur ^= 1;
  }

  for (int i = 0; i < 2; ++i) {
    int rowbase = bm + wm * 32 + i * 16 + sub * 8;
    for (int j = 0; j < 2; ++j) {
      int col = bn + wn * 32 + j * 16 + l15;
      float bb = bias[col];
      for (int r = 0; r < 8; ++r) {
        int dtok = win_to_flat(rowbase + r);       // window reverse
        Out[(size_t)dtok * C_DIM + col] = acc[i][j].s[r] + bb;
      }
    }
  }
}

// ---------------------------------------------------------------------------
extern "C" void kernel_launch(void* const* d_in, const int* in_sizes, int n_in,
                              void* d_out, int out_size, void* d_ws, size_t ws_size,
                              hipStream_t stream) {
  const float* hs = (const float*)d_in[0];
  const float* Wq = (const float*)d_in[1];
  const float* bq = (const float*)d_in[2];
  const float* Wk = (const float*)d_in[3];
  const float* bk = (const float*)d_in[4];
  const float* Wv = (const float*)d_in[5];
  const float* bv = (const float*)d_in[6];
  const float* Wo = (const float*)d_in[7];
  const float* bo = (const float*)d_in[8];
  float* out = (float*)d_out;

  char* ws = (char*)d_ws;
  const size_t SZ  = (size_t)TTOT * C_DIM * sizeof(_Float16);   // 100663296 B
  const size_t WSZ = (size_t)C_DIM * C_DIM * sizeof(_Float16);  // 1179648 B
  _Float16* Xw   = (_Float16*)(ws);            // aliased as Ctx after QKV GEMMs
  _Float16* Qb   = (_Float16*)(ws + SZ);
  _Float16* Kb   = (_Float16*)(ws + 2 * SZ);
  _Float16* Vt   = (_Float16*)(ws + 3 * SZ);
  _Float16* W16q = (_Float16*)(ws + 4 * SZ);
  _Float16* W16k = (_Float16*)(ws + 4 * SZ + WSZ);
  _Float16* W16v = (_Float16*)(ws + 4 * SZ + 2 * WSZ);
  _Float16* W16o = (_Float16*)(ws + 4 * SZ + 3 * WSZ);

  prep_xw<<<TTOT, 256, 0, stream>>>(hs, Xw);
  cvt_f32_f16<<<576, 256, 0, stream>>>(Wq, W16q, C_DIM * C_DIM);
  cvt_f32_f16<<<576, 256, 0, stream>>>(Wk, W16k, C_DIM * C_DIM);
  cvt_f32_f16<<<576, 256, 0, stream>>>(Wv, W16v, C_DIM * C_DIM);
  cvt_f32_f16<<<576, 256, 0, stream>>>(Wo, W16o, C_DIM * C_DIM);

  dim3 gg(TTOT / 128, C_DIM / 64);
  gemm_proj<0><<<gg, 256, 0, stream>>>(Xw, W16q, bq, Qb);
  gemm_proj<0><<<gg, 256, 0, stream>>>(Xw, W16k, bk, Kb);
  gemm_proj<1><<<gg, 256, 0, stream>>>(Xw, W16v, bv, Vt);

  attn<<<(NWIN * NHEAD) / 8, 256, 0, stream>>>(Qb, Kb, Vt, Xw /* Ctx alias */);

  gemm_out<<<gg, 256, 0, stream>>>(Xw /* Ctx */, W16o, bo, out);
}
